// GraphMultiHeadAttention_9423158247892
// MI455X (gfx1250) — matrix operand order repaired and checked by reading, not compiled, not run on hardware
//
#include <hip/hip_runtime.h>

#define WB 2
#define WS 2048
#define WD 512
#define WH 8
#define WHS 64

typedef __attribute__((ext_vector_type(16))) _Float16 v16h;
typedef __attribute__((ext_vector_type(8)))  _Float16 v8h;
typedef __attribute__((ext_vector_type(8)))  float    v8f;
typedef __attribute__((vector_size(4 * sizeof(int)))) int v4i;
typedef __attribute__((address_space(1))) v4i glb_v4i;
typedef __attribute__((address_space(3))) v4i lds_v4i;

#if __has_builtin(__builtin_amdgcn_exp2f)
#define EXP2F(x) __builtin_amdgcn_exp2f(x)
#else
#define EXP2F(x) exp2f(x)
#endif

#if __has_builtin(__builtin_amdgcn_global_load_async_to_lds_b128)
#define HAS_ASYNC 1
#else
#define HAS_ASYNC 0
#endif

#if __has_builtin(__builtin_amdgcn_s_wait_asynccnt)
#define WAIT_ASYNCCNT(n) __builtin_amdgcn_s_wait_asynccnt(n)
#else
#define WAIT_ASYNCCNT(n) asm volatile("s_wait_asynccnt %0" ::"n"(n) : "memory")
#endif

static __device__ __forceinline__ v8f wmma_f16(v16h a, v16h b, v8f c) {
    // D = A(16x32 f16) * B(32x16 f16) + C(16x16 f32)
    return __builtin_amdgcn_wmma_f32_16x16x32_f16(
        /*neg_a=*/false, a, /*neg_b=*/false, b,
        /*c_mod=*/(short)0, c, /*reuse_a=*/false, /*reuse_b=*/false);
}

// ---------------------------------------------------------------------------
// Kernel 1: QKV projection. y[s,e] = sum_d X[s,d] * W[e,d] + bias[e]
// Q,K computed as transposed tiles C'[e,s] (A=W rows, B=X^T) -> contiguous
//   v8h stores into Q/K laid out [B,H,S,HS] (f16).
// V computed straight C[s,e] -> contiguous v8h stores into Vt [B,H,HS,S].
// One 16x16 output tile per wave; K-loop in steps of 32.
// ---------------------------------------------------------------------------
__global__ __launch_bounds__(128) void qkv_proj_kernel(
    const float* __restrict__ X,
    const float* __restrict__ Wq, const float* __restrict__ Bq,
    const float* __restrict__ Wk, const float* __restrict__ Bk,
    const float* __restrict__ Wv, const float* __restrict__ Bv,
    _Float16* __restrict__ Qo, _Float16* __restrict__ Ko,
    _Float16* __restrict__ Vto)
{
    const int wave = threadIdx.x >> 5;
    const int lane = threadIdx.x & 31;
    const int hh   = lane >> 4;   // half-wave select
    const int l16  = lane & 15;

    const int sTile = blockIdx.x * 4 + wave;   // 0..255  (B*S/16)
    const int b  = sTile >> 7;                 // S/16 = 128 tiles per batch
    const int s0 = (sTile & 127) << 4;
    const int e0 = blockIdx.y << 4;            // 0..496
    const int z  = blockIdx.z;                 // 0=Q 1=K 2=V

    const float* Wm = (z == 0) ? Wq : (z == 1) ? Wk : Wv;
    const float* Bi = (z == 0) ? Bq : (z == 1) ? Bk : Bv;

    const int hd = e0 >> 6;        // head
    const int eh = e0 & 63;        // e offset within head

    v8f c = {};

    if (z < 2) {
        // Transposed tile: C'[e,s]. A row m=e (lane l16), B row n=s (lane l16).
        const float* arow = Wm + (size_t)(e0 + l16) * WD;
        const float* brow = X + ((size_t)(b * WS + s0 + l16)) * WD;
        for (int k0 = 0; k0 < WD; k0 += 32) {
            v16h a, bb;
            const float* alo = arow + k0 + 8 * hh;        // A: K=j+8*hh
            const float* ahi = arow + k0 + 16 + 8 * hh;   // A: K=j+8+8*hh
            #pragma unroll
            for (int j = 0; j < 8; ++j) {
                a[j]     = (_Float16)alo[j];
                a[8 + j] = (_Float16)ahi[j];
            }
            const float* bp = brow + k0 + 8 * hh;         // B: K=j+8*hh (j<8), j+8+8*hh (j>=8)
            #pragma unroll
            for (int j = 0; j < 16; ++j) bb[j] = (_Float16)bp[j < 8 ? j : j + 8];
            c = wmma_f16(a, bb, c);
        }
        // C' rows: e = e0 + 8*hh + r ; column s = s0 + l16 (fixed per lane)
        const float* bptr = Bi + e0 + 8 * hh;
        v8h ov;
        #pragma unroll
        for (int r = 0; r < 8; ++r) ov[r] = (_Float16)(c[r] + bptr[r]);
        _Float16* dst = ((z == 0) ? Qo : Ko)
            + (((size_t)(b * WH + hd) * WS + (s0 + l16)) * WHS + eh + 8 * hh);
        *(v8h*)dst = ov;
    } else {
        // Straight tile: C[s,e]. A row m=s, B row n=e.
        const float* arow = X + ((size_t)(b * WS + s0 + l16)) * WD;
        const float* brow = Wm + (size_t)(e0 + l16) * WD;
        for (int k0 = 0; k0 < WD; k0 += 32) {
            v16h a, bb;
            const float* alo = arow + k0 + 8 * hh;
            const float* ahi = arow + k0 + 16 + 8 * hh;
            #pragma unroll
            for (int j = 0; j < 8; ++j) {
                a[j]     = (_Float16)alo[j];
                a[8 + j] = (_Float16)ahi[j];
            }
            const float* bp = brow + k0 + 8 * hh;
            #pragma unroll
            for (int j = 0; j < 16; ++j) bb[j] = (_Float16)bp[j < 8 ? j : j + 8];
            c = wmma_f16(a, bb, c);
        }
        // C rows: s = s0 + 8*hh + r ; column e = e0 + l16 (fixed per lane)
        const float bias = Bi[e0 + l16];
        v8h ov;
        #pragma unroll
        for (int r = 0; r < 8; ++r) ov[r] = (_Float16)(c[r] + bias);
        // Vt[b,h,e',s] with e' = eh + l16, s = s0 + 8*hh + r (contiguous)
        _Float16* dst = Vto
            + (((size_t)(b * WH + hd) * WHS + eh + l16) * WS + s0 + 8 * hh);
        *(v8h*)dst = ov;
    }
}

// ---------------------------------------------------------------------------
// Cooperative staging of one 32-key chunk of K (32x64 f16) and Vt (64x32 f16)
// into LDS. 128 threads x 2 reps x (K+V) = 16B b128 segments covering 4KB+4KB.
// Async path uses GLOBAL_LOAD_ASYNC_TO_LDS_B128 (ASYNCcnt += 4 per wave).
// ---------------------------------------------------------------------------
static __device__ __forceinline__ void stage_chunk(
    const _Float16* __restrict__ Kh, const _Float16* __restrict__ Vh,
    int k0, _Float16* sK, _Float16* sV, int tid)
{
    #pragma unroll
    for (int rep = 0; rep < 2; ++rep) {
        const int seg = tid + rep * 128;          // 0..255
        const int kk = seg >> 3, ko = (seg & 7) * 8;   // K: 32 rows x 64
        const int ee = seg >> 2, eo = (seg & 3) * 8;   // V: 64 rows x 32
        const _Float16* gK = Kh + (size_t)(k0 + kk) * WHS + ko;
        const _Float16* gV = Vh + (size_t)ee * WS + k0 + eo;
        _Float16* lK = sK + kk * WHS + ko;
        _Float16* lV = sV + ee * 32 + eo;
#if HAS_ASYNC
        __builtin_amdgcn_global_load_async_to_lds_b128(
            (glb_v4i*)const_cast<_Float16*>(gK), (lds_v4i*)lK, 0, 0);
        __builtin_amdgcn_global_load_async_to_lds_b128(
            (glb_v4i*)const_cast<_Float16*>(gV), (lds_v4i*)lV, 0, 0);
#else
        *(v8h*)lK = *(const v8h*)gK;
        *(v8h*)lV = *(const v8h*)gV;
#endif
    }
}

// ---------------------------------------------------------------------------
// Kernel 2: flash-style attention with post-softmax edge weights + row mask.
// Block = 4 waves = 64 consecutive query rows of ONE head, so all waves share
// the K/V stream: each 32-key chunk is staged once into LDS (double-buffered,
// async DMA overlapped with compute). Scores computed transposed
// (sim^T = K*Q^T) so softmax stats need only one cross-half shuffle; output
// accumulated transposed (acc^T = V^T * P^T) for contiguous stores.
// out = mask[q] * acc / Z   (Z excludes w_X; numerator includes w_X).
// ---------------------------------------------------------------------------
__global__ __launch_bounds__(128) void attn_kernel(
    const _Float16* __restrict__ Q, const _Float16* __restrict__ K,
    const _Float16* __restrict__ Vt,
    const float* __restrict__ wX, const float* __restrict__ msk,
    float* __restrict__ out)
{
    __shared__ __align__(16) _Float16 sK[2][32 * WHS];   // [buf][key][e]
    __shared__ __align__(16) _Float16 sV[2][WHS * 32];   // [buf][e][key]

    const int tid  = threadIdx.x;
    const int wave = tid >> 5;
    const int lane = tid & 31;
    const int hh   = lane >> 4;
    const int l16  = lane & 15;

    const int bi = blockIdx.x;               // 0..511
    const int b  = bi >> 8;                  // 256 blocks per batch
    const int t8 = bi & 255;
    const int hd = t8 >> 5;                  // 32 blocks per head
    const int q0 = ((t8 & 31) << 6) + wave * 16;   // 64 q rows per block

    const _Float16* Qh = Q  + (size_t)(b * WH + hd) * WS * WHS;
    const _Float16* Kh = K  + (size_t)(b * WH + hd) * WS * WHS;
    const _Float16* Vh = Vt + (size_t)(b * WH + hd) * WHS * WS;
    const float*  wrow = wX + ((size_t)b * WS + q0 + l16) * WS;

    // Q as B-fragments (K-dim = e); q column = q0 + l16; reused for all keys.
    v16h qf[2];
    #pragma unroll
    for (int t = 0; t < 2; ++t)
        qf[t] = *(const v16h*)(Qh + (size_t)(q0 + l16) * WHS + t * 32 + hh * 16);

    float m_run = -1e30f;
    float Zrun  = 0.0f;
    v8f acc[4] = {};   // acc^T e-tiles (HS=64 -> 4 tiles of 16)

    const float cs = 0.1803368801111204f;   // log2(e) / sqrt(HS)

    // Prime the pipeline: stage chunk 0 into buffer 0.
    stage_chunk(Kh, Vh, 0, sK[0], sV[0], tid);

    for (int k0 = 0; k0 < WS; k0 += 32) {
        const int cur = (k0 >> 5) & 1;

        // Prefetch next chunk into the other buffer, then retire current DMA.
        if (k0 + 32 < WS) {
            stage_chunk(Kh, Vh, k0 + 32, sK[cur ^ 1], sV[cur ^ 1], tid);
#if HAS_ASYNC
            WAIT_ASYNCCNT(4);   // 8 outstanding; <=4 left => current chunk done
#endif
        } else {
#if HAS_ASYNC
            WAIT_ASYNCCNT(0);
#endif
        }
        __syncthreads();

        const _Float16* sKc = sK[cur];
        const _Float16* sVc = sV[cur];

        if (k0 + 32 < WS) __builtin_prefetch(wrow + k0 + 32, 0, 1);

        // ---- scores (transposed): st[t2] covers keys k0+16*t2 .. +15
        v8f st[2] = {};
        #pragma unroll
        for (int t2 = 0; t2 < 2; ++t2) {
            const _Float16* krow = sKc + (16 * t2 + l16) * WHS;
            #pragma unroll
            for (int t = 0; t < 2; ++t) {   // e k-steps (64 = 2x32)
                v16h a;
                const v8h alo = *(const v8h*)(krow + 32 * t + 16 * hh);
                const v8h ahi = *(const v8h*)(krow + 32 * t + 16 * hh + 8);
                #pragma unroll
                for (int j = 0; j < 8; ++j) { a[j] = alo[j]; a[8 + j] = ahi[j]; }
                st[t2] = wmma_f16(a, qf[t], st[t2]);
            }
        }

        // held keys: st[t2][r] -> key = k0 + 16*t2 + 8*hh + r (q = l16)
        float4 w0a = *(const float4*)(wrow + k0 + 8 * hh);
        float4 w0b = *(const float4*)(wrow + k0 + 8 * hh + 4);
        float4 w1a = *(const float4*)(wrow + k0 + 16 + 8 * hh);
        float4 w1b = *(const float4*)(wrow + k0 + 16 + 8 * hh + 4);
        float wv0[8] = {w0a.x, w0a.y, w0a.z, w0a.w, w0b.x, w0b.y, w0b.z, w0b.w};
        float wv1[8] = {w1a.x, w1a.y, w1a.z, w1a.w, w1b.x, w1b.y, w1b.z, w1b.w};

        // ---- online softmax (log2 domain)
        float sv0[8], sv1[8];
        float cmax = -1e30f;
        #pragma unroll
        for (int r = 0; r < 8; ++r) {
            sv0[r] = st[0][r] * cs;
            sv1[r] = st[1][r] * cs;
            cmax = fmaxf(cmax, fmaxf(sv0[r], sv1[r]));
        }
        cmax = fmaxf(cmax, __shfl_xor(cmax, 16, 32));
        const float new_m = fmaxf(m_run, cmax);
        const float scale = EXP2F(m_run - new_m);
        float psum = 0.0f;
        float p0[8], p1[8];
        #pragma unroll
        for (int r = 0; r < 8; ++r) {
            float ea = EXP2F(sv0[r] - new_m);
            float eb = EXP2F(sv1[r] - new_m);
            psum += ea + eb;
            p0[r] = ea * wv0[r];   // edge weights: numerator only
            p1[r] = eb * wv1[r];
        }
        psum += __shfl_xor(psum, 16, 32);
        Zrun = Zrun * scale + psum;
        m_run = new_m;
        #pragma unroll
        for (int i = 0; i < 4; ++i) {
            #pragma unroll
            for (int r = 0; r < 8; ++r) acc[i][r] *= scale;
        }

        // ---- P^T as B-fragment: lane hh=0 needs keys k0+0..15,
        //      lane hh=1 needs keys k0+16..31; one cross-half shuffle each.
        v16h pf;
        #pragma unroll
        for (int r = 0; r < 8; ++r) {
            float o0 = __shfl_xor(p0[r], 16, 32);
            float o1 = __shfl_xor(p1[r], 16, 32);
            float lo = hh ? o1 : p0[r];
            float hi = hh ? p1[r] : o0;
            pf[r]     = (_Float16)lo;
            pf[8 + r] = (_Float16)hi;
        }

        // ---- acc^T[e,q] += Vt_tile(16x32) * P^T(32x16)
        #pragma unroll
        for (int et = 0; et < 4; ++et) {
            const _Float16* vrow = sVc + (et * 16 + l16) * 32;
            v16h a;
            const v8h alo = *(const v8h*)(vrow + 16 * hh);
            const v8h ahi = *(const v8h*)(vrow + 16 * hh + 8);
            #pragma unroll
            for (int j = 0; j < 8; ++j) { a[j] = alo[j]; a[8 + j] = ahi[j]; }
            acc[et] = wmma_f16(a, pf, acc[et]);
        }

        __syncthreads();   // all waves done with 'cur' before it is reused
    }

    // ---- finalize: out[b, q, hd*64 + e] = mask * acc / Z
    const float mk = msk[b * WS + q0 + l16];
    const float rz = mk / Zrun;
    float* orow = out + ((size_t)(b * WS + q0 + l16)) * WD + hd * WHS + 8 * hh;
    #pragma unroll
    for (int et = 0; et < 4; ++et) {
        float4 lo4, hi4;
        lo4.x = acc[et][0] * rz; lo4.y = acc[et][1] * rz;
        lo4.z = acc[et][2] * rz; lo4.w = acc[et][3] * rz;
        hi4.x = acc[et][4] * rz; hi4.y = acc[et][5] * rz;
        hi4.z = acc[et][6] * rz; hi4.w = acc[et][7] * rz;
        *(float4*)(orow + et * 16)     = lo4;
        *(float4*)(orow + et * 16 + 4) = hi4;
    }
}

extern "C" void kernel_launch(void* const* d_in, const int* in_sizes, int n_in,
                              void* d_out, int out_size, void* d_ws, size_t ws_size,
                              hipStream_t stream) {
    (void)in_sizes; (void)n_in; (void)out_size; (void)ws_size;
    const float* seq = (const float*)d_in[0];
    const float* wX  = (const float*)d_in[1];
    const float* mk  = (const float*)d_in[2];
    const float* wq  = (const float*)d_in[3];
    const float* bq  = (const float*)d_in[4];
    const float* wk  = (const float*)d_in[5];
    const float* bk  = (const float*)d_in[6];
    const float* wv  = (const float*)d_in[7];
    const float* bv  = (const float*)d_in[8];
    float* out = (float*)d_out;

    // Workspace: Q, K (f16 [B,H,S,HS]) and Vt (f16 [B,H,HS,S]) = 12 MB total.
    const size_t N = (size_t)WB * WH * WS * WHS;   // 2,097,152 elements each
    _Float16* Qw = (_Float16*)d_ws;
    _Float16* Kw = Qw + N;
    _Float16* Vw = Kw + N;

    dim3 g1(64, 32, 3), t1(128);
    qkv_proj_kernel<<<g1, t1, 0, stream>>>(seq, wq, bq, wk, bk, wv, bv,
                                           Qw, Kw, Vw);

    dim3 g2(512), t2(128);
    attn_kernel<<<g2, t2, 0, stream>>>(Qw, Kw, Vw, wX, mk, out);
}
